// PerceiverAttention_12945031430317
// MI455X (gfx1250) — matrix operand order repaired and checked by reading, not compiled, not run on hardware
//
#include <hip/hip_runtime.h>

// ---------------------------------------------------------------------------
// PerceiverAttention for MI455X (gfx1250): bf16 WMMA pipeline, fp32 softmax/LN
// Double-buffered TDM (tensor_load_to_lds) tile staging overlapping DMA with
// WMMA compute; manual-staging fallback.
// B=8 T=4 N1=1024 N2=64 DIM=1024 HEADS=8 DHEAD=64 INNER=512 -> BT=32 batches
// ---------------------------------------------------------------------------

typedef __bf16 bf16;
typedef __attribute__((ext_vector_type(16))) __bf16 v16bf;
typedef __attribute__((ext_vector_type(8)))  __bf16 v8bf;
typedef __attribute__((ext_vector_type(8)))  float  v8f;
typedef unsigned int u32x4 __attribute__((ext_vector_type(4)));
typedef int i32x4 __attribute__((ext_vector_type(4)));
typedef int i32x8 __attribute__((ext_vector_type(8)));

#if defined(__has_builtin)
#if __has_builtin(__builtin_amdgcn_tensor_load_to_lds) && \
    __has_builtin(__builtin_amdgcn_s_wait_tensorcnt)
#define USE_TDM 1
#endif
#endif
#ifndef USE_TDM
#define USE_TDM 0
#endif

__device__ __forceinline__ v8f wmma_bf16(v16bf a, v16bf b, v8f c) {
  return __builtin_amdgcn_wmma_f32_16x16x32_bf16(false, a, false, b,
                                                 (short)0, c, false, false);
}

__device__ __forceinline__ v8f zero8() {
  v8f z = {0.f, 0.f, 0.f, 0.f, 0.f, 0.f, 0.f, 0.f};
  return z;
}

#if USE_TDM
// ---------------------------------------------------------------------------
// TDM: DMA a 2D tile (tileRows x tileCols bf16, row stride rowStrideElems)
// from global memory into LDS at byte offset ldsAddr.  Optional LDS padding:
// after every (2<<padInterval) DWORDs stored, skip padDwords DWORDs (gives a
// conflict-free padded LDS stride for free).
// D# bitfields per CDNA5 ISA.
// ---------------------------------------------------------------------------
__device__ __forceinline__ void tdm_load_2d(unsigned ldsAddr, const void* gaddr,
                                            unsigned tileCols, unsigned tileRows,
                                            unsigned rowStrideElems,
                                            unsigned padInterval,
                                            unsigned padDwords) {
  unsigned long long ga = (unsigned long long)(size_t)gaddr;
  u32x4 g0;
  g0.x = 1u;                                    // count=1 (valid), user mode
  g0.y = ldsAddr;                               // lds_addr [63:32]
  g0.z = (unsigned)ga;                          // global_addr [95:64]
  g0.w = (unsigned)((ga >> 32) & 0x01ffffffu)   // global_addr [120:96]
         | (2u << 30);                          // type=2 ("image")
  i32x8 g1;
  unsigned w0 = (1u << 16);                     // data_size=1 (2 bytes)
  if (padDwords)
    w0 |= (1u << 20)                            // pad_enable
        | (padInterval << 22)                   // pad_interval
        | ((padDwords - 1u) << 25);             // pad_amount (0 => 1 DWORD)
  g1[0] = (int)w0;                              // wg_mask=0 (no cluster)
  g1[1] = (int)((tileCols & 0xffffu) << 16);    // tensor_dim0[15:0] @ bits63:48
  g1[2] = (int)((tileCols >> 16) | ((tileRows & 0xffffu) << 16)); // d0 hi|d1 lo
  g1[3] = (int)((tileRows >> 16) | ((tileCols & 0xffffu) << 16)); // d1 hi|tile_dim0
  g1[4] = (int)(tileRows & 0xffffu);            // tile_dim1 (tile_dim2 = 0)
  g1[5] = (int)rowStrideElems;                  // tensor_dim0_stride[31:0]
  g1[6] = 0;                                    // stride hi / dim1_stride lo
  g1[7] = 0;
  i32x4 z4 = {0, 0, 0, 0};
#if __clang_major__ >= 23
  i32x8 z8 = {0, 0, 0, 0, 0, 0, 0, 0};
  __builtin_amdgcn_tensor_load_to_lds(g0, g1, z4, z4, z8, 0);
#else
  __builtin_amdgcn_tensor_load_to_lds(g0, g1, z4, z4, 0);
#endif
}
#define TDM_WAIT() __builtin_amdgcn_s_wait_tensorcnt(0)
#endif  // USE_TDM

// A-fragment (16x32 bf16, MxK): lane L holds row m=L&15.
__device__ __forceinline__ v16bf ldAfrag(const bf16* rowk0, int lane) {
  int kh = (lane >> 4) * 8;
  v8bf lo = *(const v8bf*)(rowk0 + kh);
  v8bf hi = *(const v8bf*)(rowk0 + kh + 16);
  return __builtin_shufflevector(lo, hi, 0,1,2,3,4,5,6,7,8,9,10,11,12,13,14,15);
}

// Same fragment but fp32 source (attention probabilities in LDS)
__device__ __forceinline__ v16bf ldAfrag_f32(const float* rowk0, int lane) {
  int kh = (lane >> 4) * 8;
  const float* p = rowk0 + kh;
  v16bf a;
#pragma unroll
  for (int e = 0; e < 8; ++e) {
    a[e]     = (bf16)p[e];
    a[e + 8] = (bf16)p[e + 16];
  }
  return a;
}

// B-fragment (32x16 bf16, KxN): lane L holds column n=L&15, K-contiguous src.
__device__ __forceinline__ v16bf ldBfrag(const bf16* colk0, int lane) {
  int kb = (lane >> 4) * 8;
  v8bf lo = *(const v8bf*)(colk0 + kb);
  v8bf hi = *(const v8bf*)(colk0 + kb + 16);
  return __builtin_shufflevector(lo, hi, 0,1,2,3,4,5,6,7,8,9,10,11,12,13,14,15);
}

// ---------------------------------------------------------------------------
// Kernel 1: LayerNorm (fp32) -> bf16 into concatenated (bt,1088,1024) buffer.
// ---------------------------------------------------------------------------
__global__ void ln_cast_kernel(const float* __restrict__ in,
                               const float* __restrict__ g,
                               const float* __restrict__ bparm,
                               bf16* __restrict__ out,
                               int rowsPerBT, int outRowOff) {
  int row = blockIdx.x;
  int bt  = row / rowsPerBT;
  int i   = row - bt * rowsPerBT;
  int t = threadIdx.x, lane = t & 31, w = t >> 5;

  const float* x = in + (long long)row * 1024 + t * 4;
  float4 v = *(const float4*)x;
  float s1 = v.x + v.y + v.z + v.w;
  float s2 = v.x * v.x + v.y * v.y + v.z * v.z + v.w * v.w;
#pragma unroll
  for (int off = 16; off > 0; off >>= 1) {
    s1 += __shfl_down(s1, off, 32);
    s2 += __shfl_down(s2, off, 32);
  }
  __shared__ float red1[8], red2[8], stat[2];
  if (lane == 0) { red1[w] = s1; red2[w] = s2; }
  __syncthreads();
  if (t == 0) {
    float a = 0.f, b2 = 0.f;
#pragma unroll
    for (int k = 0; k < 8; ++k) { a += red1[k]; b2 += red2[k]; }
    float mean = a * (1.0f / 1024.0f);
    float var  = b2 * (1.0f / 1024.0f) - mean * mean;
    stat[0] = mean;
    stat[1] = rsqrtf(var + 1e-5f);
  }
  __syncthreads();
  float mean = stat[0], rstd = stat[1];
  float4 gg = *(const float4*)(g + t * 4);
  float4 bb = *(const float4*)(bparm + t * 4);
  bf16* o = out + ((long long)bt * 1088 + outRowOff + i) * 1024 + t * 4;
  o[0] = (bf16)((v.x - mean) * rstd * gg.x + bb.x);
  o[1] = (bf16)((v.y - mean) * rstd * gg.y + bb.y);
  o[2] = (bf16)((v.z - mean) * rstd * gg.z + bb.z);
  o[3] = (bf16)((v.w - mean) * rstd * gg.w + bb.w);
}

// ---------------------------------------------------------------------------
// Kernel 2: weight fp32 (K x N) -> bf16 transposed (N x K)
// ---------------------------------------------------------------------------
__global__ void castT_kernel(const float* __restrict__ W, bf16* __restrict__ WT,
                             int K, int N) {
  long long idx = (long long)blockIdx.x * blockDim.x + threadIdx.x;
  long long total = (long long)K * N;
  if (idx >= total) return;
  long long k = idx / N, n = idx - k * N;
  WT[n * K + k] = (bf16)W[k * N + n];
}

// ---------------------------------------------------------------------------
// Kernel 3: batched GEMM  C[b] = A[b] (MxK bf16) x B (KxN), B given as B^T.
// Block tile 64x128, BK=32; 8 waves x 4 wmma tiles.
// Double-buffered TDM staging: DMA of tile kt+1 overlaps WMMA on tile kt.
// ---------------------------------------------------------------------------
template <bool OUT_BF16>
__global__ void gemm_bt_kernel(const bf16* __restrict__ A,
                               const bf16* __restrict__ BT,
                               void* __restrict__ C,
                               int N, int K, int lda, int ldc,
                               long long strideA, long long strideC) {
  const int tilesN = N >> 7;
  const int tm = blockIdx.x / tilesN;
  const int tn = blockIdx.x - tm * tilesN;
  const int batch = blockIdx.y;
  const int t = threadIdx.x, lane = t & 31, w = t >> 5;

  __shared__ __align__(16) bf16 lA[2][64 * 40];   // padded stride 40 (80B rows)
  __shared__ __align__(16) bf16 lB[2][128 * 40];

  const bf16* Ab = A + (long long)batch * strideA;
  const int mt  = w & 3;
  const int ntb = (w >> 2) * 4;

  v8f acc[4] = {zero8(), zero8(), zero8(), zero8()};
  const int kTiles = K >> 5;

#if USE_TDM
  const unsigned ldsA[2] = {(unsigned)(size_t)&lA[0][0],
                            (unsigned)(size_t)&lA[1][0]};
  const unsigned ldsB[2] = {(unsigned)(size_t)&lB[0][0],
                            (unsigned)(size_t)&lB[1][0]};
  const bf16* aTile = Ab + (long long)(tm * 64) * lda;
  const bf16* bTile = BT + (long long)(tn * 128) * K;

  // prologue: DMA first tiles into buffer 0
  if (w == 0)      tdm_load_2d(ldsA[0], aTile, 32u, 64u, (unsigned)lda, 3u, 4u);
  else if (w == 1) tdm_load_2d(ldsB[0], bTile, 32u, 128u, (unsigned)K, 3u, 4u);
  TDM_WAIT();
  __syncthreads();

  for (int kt = 0; kt < kTiles; ++kt) {
    const int cur = kt & 1, nxt = cur ^ 1;
    if (kt + 1 < kTiles) {  // DMA next K-slab while we compute this one
      if (w == 0)
        tdm_load_2d(ldsA[nxt], aTile + (kt + 1) * 32, 32u, 64u, (unsigned)lda,
                    3u, 4u);
      else if (w == 1)
        tdm_load_2d(ldsB[nxt], bTile + (kt + 1) * 32, 32u, 128u, (unsigned)K,
                    3u, 4u);
      if (kt + 2 < kTiles)  // gfx1250 global_prefetch_b8 for the slab after
        __builtin_prefetch(aTile + (kt + 2) * 32 + (long long)(t >> 2) * lda,
                           0, 0);
    }

    v16bf af = ldAfrag(&lA[cur][(mt * 16 + (lane & 15)) * 40], lane);
#pragma unroll
    for (int i = 0; i < 4; ++i) {
      v16bf bfr = ldBfrag(&lB[cur][((ntb + i) * 16 + (lane & 15)) * 40], lane);
      acc[i] = wmma_bf16(af, bfr, acc[i]);
    }

    TDM_WAIT();        // issuing waves: next tile landed; others: no-op
    __syncthreads();   // all waves done reading 'cur' before it is reused
  }
#else
  const int arow = t >> 2, aseg = t & 3;
  const int brow = t >> 1, bseg = t & 1;
  const bf16* aSrc = Ab + (long long)(tm * 64 + arow) * lda + aseg * 8;
  const bf16* bSrc = BT + (long long)(tn * 128 + brow) * K + bseg * 16;
  for (int kt = 0; kt < kTiles; ++kt) {
    *(v8bf*)(&lA[0][arow * 40 + aseg * 8])      = *(const v8bf*)(aSrc);
    *(v8bf*)(&lB[0][brow * 40 + bseg * 16])     = *(const v8bf*)(bSrc);
    *(v8bf*)(&lB[0][brow * 40 + bseg * 16 + 8]) = *(const v8bf*)(bSrc + 8);
    aSrc += 32;
    bSrc += 32;
    if (kt + 1 < kTiles) {
      __builtin_prefetch(aSrc, 0, 0);
      __builtin_prefetch(bSrc, 0, 0);
    }
    __syncthreads();
    v16bf af = ldAfrag(&lA[0][(mt * 16 + (lane & 15)) * 40], lane);
#pragma unroll
    for (int i = 0; i < 4; ++i) {
      v16bf bfr = ldBfrag(&lB[0][((ntb + i) * 16 + (lane & 15)) * 40], lane);
      acc[i] = wmma_bf16(af, bfr, acc[i]);
    }
    __syncthreads();
  }
#endif

  const long long cb = (long long)batch * strideC;
#pragma unroll
  for (int i = 0; i < 4; ++i) {
    int col = tn * 128 + (ntb + i) * 16 + (lane & 15);
    int mr  = tm * 64 + mt * 16 + (lane >> 4) * 8;
#pragma unroll
    for (int r = 0; r < 8; ++r) {
      long long off = cb + (long long)(mr + r) * ldc + col;
      if (OUT_BF16) ((bf16*)C)[off] = (bf16)acc[i][r];
      else          ((float*)C)[off] = acc[i][r];
    }
  }
}

// ---------------------------------------------------------------------------
// Kernel 4: attention for one (bt, head). Full 64x1088 fp32 score matrix in
// LDS (320KB WGP LDS). WMMA for S=q@k^T and O=P@V.
// Phase 1 uses double-buffered TDM for the k tiles.
// ---------------------------------------------------------------------------
#define S_STRIDE 1104   // 1088 + 16 pad (floats)
// scores + q tile + 2x k tile (both k buffers double as v^T scratch space)
#define ATTN_SMEM (64 * S_STRIDE * 4 + 64 * 64 * 2 + 2 * 64 * 64 * 2) // 307200

__global__ void attn_kernel(const bf16* __restrict__ qb,
                            const bf16* __restrict__ kvb,
                            bf16* __restrict__ ob) {
  const int h = blockIdx.x, bt = blockIdx.y;
  const int t = threadIdx.x, lane = t & 31, w = t >> 5;

  extern __shared__ char smem[];
  float* lS   = (float*)smem;                        // 64 x S_STRIDE fp32
  bf16*  lQ   = (bf16*)(smem + 64 * S_STRIDE * 4);   // 64 x 64
  bf16*  lKV0 = lQ + 64 * 64;                        // k tile buf 0
  bf16*  lKV1 = lKV0 + 64 * 64;                      // k tile buf 1
  // phase 3 reuses lKV0.. as v^T (64 x 72 = 9216B < 16384B available)

  const int mt = w & 3;          // wave's M-tile (of 4)
  const int pb = (w >> 2) * 2;   // wave's pair base among 4 N-tiles

  const bf16* kBase = kvb + ((long long)bt * 1088) * 1024 + h * 64;

  // ---- phase 1: S = scale * q @ k^T, 17 j-tiles of 64, double-buffered ----
#if USE_TDM
  const unsigned ldsQ = (unsigned)(size_t)lQ;
  const unsigned ldsK[2] = {(unsigned)(size_t)lKV0, (unsigned)(size_t)lKV1};
  if (w == 0) {
    tdm_load_2d(ldsQ, qb + ((long long)bt * 64) * 512 + h * 64, 64u, 64u, 512u,
                0u, 0u);
    tdm_load_2d(ldsK[0], kBase, 64u, 64u, 1024u, 0u, 0u);
  }
  TDM_WAIT();
  __syncthreads();
  for (int jt = 0; jt < 17; ++jt) {
    const int cur = jt & 1;
    bf16* kTile = cur ? lKV1 : lKV0;
    if (jt + 1 < 17 && w == 0)
      tdm_load_2d(ldsK[cur ^ 1], kBase + (long long)(jt + 1) * 64 * 1024, 64u,
                  64u, 1024u, 0u, 0u);
#else
  {
    int m = t >> 2, seg = t & 3;
    const bf16* src = qb + ((long long)bt * 64 + m) * 512 + h * 64 + seg * 16;
    *(v8bf*)(lQ + m * 64 + seg * 16)     = *(const v8bf*)src;
    *(v8bf*)(lQ + m * 64 + seg * 16 + 8) = *(const v8bf*)(src + 8);
  }
  for (int jt = 0; jt < 17; ++jt) {
    bf16* kTile = lKV0;
    {
      int j = t >> 2, seg = t & 3;
      const bf16* src = kBase + (long long)(jt * 64 + j) * 1024 + seg * 16;
      *(v8bf*)(lKV0 + j * 64 + seg * 16)     = *(const v8bf*)src;
      *(v8bf*)(lKV0 + j * 64 + seg * 16 + 8) = *(const v8bf*)(src + 8);
    }
    __syncthreads();
#endif
#pragma unroll
    for (int j2 = 0; j2 < 2; ++j2) {
      v8f acc = zero8();
#pragma unroll
      for (int ks = 0; ks < 2; ++ks) {
        v16bf a = ldAfrag(lQ + (mt * 16 + (lane & 15)) * 64 + ks * 32, lane);
        v16bf b = ldBfrag(kTile + ((pb + j2) * 16 + (lane & 15)) * 64 + ks * 32,
                          lane);
        acc = wmma_bf16(a, b, acc);
      }
      int col = jt * 64 + (pb + j2) * 16 + (lane & 15);
      int mr  = mt * 16 + (lane >> 4) * 8;
#pragma unroll
      for (int r = 0; r < 8; ++r)
        lS[(mr + r) * S_STRIDE + col] = acc[r] * 0.125f;  // DHEAD^-0.5
    }
#if USE_TDM
    TDM_WAIT();
#endif
    __syncthreads();
  }

  // ---- phase 2: row softmax over 1088, 4 threads per row ----
  {
    int r = t >> 2, q4 = t & 3;
    float* row = lS + r * S_STRIDE + q4 * 272;
    float mx = -3.4e38f;
    for (int c = 0; c < 272; ++c) mx = fmaxf(mx, row[c]);
    mx = fmaxf(mx, __shfl_xor(mx, 1, 32));
    mx = fmaxf(mx, __shfl_xor(mx, 2, 32));
    float s = 0.f;
    for (int c = 0; c < 272; ++c) {
      float e = __expf(row[c] - mx);
      row[c] = e;
      s += e;
    }
    s += __shfl_xor(s, 1, 32);
    s += __shfl_xor(s, 2, 32);
    float inv = 1.0f / s;
    for (int c = 0; c < 272; ++c) row[c] *= inv;
  }
  __syncthreads();

  // ---- phase 3: O = P @ V, V staged transposed (d-major) in LDS ----
  v8f oacc[2] = {zero8(), zero8()};
  for (int jc = 0; jc < 17; ++jc) {
    {
      int j = t >> 2, seg = t & 3;
      const bf16* src = kvb + ((long long)bt * 1088 + jc * 64 + j) * 1024 +
                        512 + h * 64 + seg * 16;
      v8bf s0 = *(const v8bf*)src;
      v8bf s1 = *(const v8bf*)(src + 8);
#pragma unroll
      for (int e = 0; e < 8; ++e) {
        lKV0[(seg * 16 + e) * 72 + j]     = s0[e];   // lVT[d][j], stride 72
        lKV0[(seg * 16 + 8 + e) * 72 + j] = s1[e];
      }
    }
    __syncthreads();
#pragma unroll
    for (int ks = 0; ks < 2; ++ks) {
      v16bf a = ldAfrag_f32(
          lS + (mt * 16 + (lane & 15)) * S_STRIDE + jc * 64 + ks * 32, lane);
#pragma unroll
      for (int d2 = 0; d2 < 2; ++d2) {
        v16bf b = ldBfrag(lKV0 + ((pb + d2) * 16 + (lane & 15)) * 72 + ks * 32,
                          lane);
        oacc[d2] = wmma_bf16(a, b, oacc[d2]);
      }
    }
    __syncthreads();
  }

  // ---- epilogue ----
#pragma unroll
  for (int d2 = 0; d2 < 2; ++d2) {
    int dcol = (pb + d2) * 16 + (lane & 15);
    int mr   = mt * 16 + (lane >> 4) * 8;
#pragma unroll
    for (int r = 0; r < 8; ++r)
      ob[((long long)bt * 64 + mr + r) * 512 + h * 64 + dcol] = (bf16)oacc[d2][r];
  }
}

// ---------------------------------------------------------------------------
// Launch
// ---------------------------------------------------------------------------
extern "C" void kernel_launch(void* const* d_in, const int* in_sizes, int n_in,
                              void* d_out, int out_size, void* d_ws,
                              size_t ws_size, hipStream_t stream) {
  (void)in_sizes; (void)n_in; (void)out_size; (void)ws_size;

  const float* x       = (const float*)d_in[0];
  const float* latents = (const float*)d_in[1];
  const float* g_media = (const float*)d_in[2];
  const float* b_media = (const float*)d_in[3];
  const float* g_lat   = (const float*)d_in[4];
  const float* b_lat   = (const float*)d_in[5];
  const float* Wq      = (const float*)d_in[6];
  const float* Wkv     = (const float*)d_in[7];
  const float* Wout    = (const float*)d_in[8];
  float* out = (float*)d_out;

  bf16* lnCat = (bf16*)d_ws;                         // 32 x 1088 x 1024
  bf16* kvb   = lnCat + (size_t)32 * 1088 * 1024;    // 32 x 1088 x 1024 (k|v)
  bf16* qb    = kvb   + (size_t)32 * 1088 * 1024;    // 32 x 64 x 512
  bf16* ob    = qb    + (size_t)32 * 64 * 512;       // 32 x 64 x 512
  bf16* WqT   = ob    + (size_t)32 * 64 * 512;       // 512 x 1024
  bf16* WkvT  = WqT   + (size_t)512 * 1024;          // 1024 x 1024
  bf16* WoutT = WkvT  + (size_t)1024 * 1024;         // 1024 x 512

  ln_cast_kernel<<<32768, 256, 0, stream>>>(x, g_media, b_media, lnCat, 1024, 0);
  ln_cast_kernel<<<2048, 256, 0, stream>>>(latents, g_lat, b_lat, lnCat, 64, 1024);

  castT_kernel<<<(1024 * 512) / 256, 256, 0, stream>>>(Wq, WqT, 1024, 512);
  castT_kernel<<<(1024 * 1024) / 256, 256, 0, stream>>>(Wkv, WkvT, 1024, 1024);
  castT_kernel<<<(512 * 1024) / 256, 256, 0, stream>>>(Wout, WoutT, 512, 1024);

  gemm_bt_kernel<true><<<dim3(17 * 8, 32), 256, 0, stream>>>(
      lnCat, WkvT, kvb, 1024, 1024, 1024, 1024,
      (long long)1088 * 1024, (long long)1088 * 1024);

  gemm_bt_kernel<true><<<dim3(1 * 4, 32), 256, 0, stream>>>(
      lnCat + (size_t)1024 * 1024, WqT, qb, 512, 1024, 1024, 512,
      (long long)1088 * 1024, (long long)64 * 512);

  attn_kernel<<<dim3(8, 32), 256, ATTN_SMEM, stream>>>(qb, kvb, ob);

  gemm_bt_kernel<false><<<dim3(1 * 8, 32), 256, 0, stream>>>(
      ob, WoutT, out, 1024, 512, 512, 1024,
      (long long)64 * 512, (long long)64 * 1024);
}
